// MultiModalEncoder_7687991460537
// MI455X (gfx1250) — matrix operand order repaired and checked by reading, not compiled, not run on hardware
//
#include <hip/hip_runtime.h>
#include <math.h>

typedef __attribute__((ext_vector_type(16))) _Float16 v16h;
typedef __attribute__((ext_vector_type(8)))  float    v8f;

#define L_   4
#define D_   512
#define FF_  2048
#define NH_  8
#define DH_  64
#define V_   32000
#define B_   8
#define S_   512
#define NN_  50000
#define NP_  50048      /* NN_ padded to a multiple of 128 for guard-free GEMM */
#define EE_  800000
#define FG_  64
#define HG_  256
#define FUS_ 768

union V16H { v16h v; unsigned u[8]; };
union H8   { _Float16 h[8]; uint4 u; };
union H4   { _Float16 h[4]; uint2 u; };

// ============================================================================
// Strided-batched WMMA GEMM (guard-free: requires M%128==0, N%32==0, K%32==0)
//   C[row,col] = act( alpha * sum_k A[row,k]*B'[k,col] + bias[col] + add[row,col] )
//   B'[k,col] = TRANSB ? Bp[col*ldb+k] : Bp[k*ldb+col]
// Block = 256 threads = 8 waves; block tile 128(M) x 32(N);
// wave w owns rows [w*16, w*16+16) and both 16-wide column tiles (2 WMMA/step).
// TRANSB is a template parameter so the inner loop is branch-free.
// ============================================================================
template<int TRANSB>
__global__ __launch_bounds__(256) void gemm_wmma_kernel(
    const float* __restrict__ A, const float* __restrict__ Bp,
    const float* __restrict__ bias, const float* __restrict__ add,
    float* __restrict__ C,
    int K, int lda, int ldb, int ldc,
    int act, float alpha,
    int zdiv, long long sA1, long long sA2, long long sB1, long long sB2,
    long long sC1, long long sC2)
{
    __shared__ _Float16 As[128][32];   // [m][k]  8KB
    __shared__ _Float16 Bs[32][32];    // [n][k]  2KB
    int z = blockIdx.z;
    long long offA = (long long)(z / zdiv) * sA1 + (long long)(z % zdiv) * sA2;
    long long offB = (long long)(z / zdiv) * sB1 + (long long)(z % zdiv) * sB2;
    long long offC = (long long)(z / zdiv) * sC1 + (long long)(z % zdiv) * sC2;

    int tid  = threadIdx.x;
    int lane = tid & 31;
    int w    = tid >> 5;              // wave id: M strip
    int bm   = blockIdx.y * 128;
    int bn   = blockIdx.x * 32;
    int r    = lane & 15;
    int hf   = lane >> 4;

    // A loader coords: thread covers one 16-wide half-row (64B contiguous)
    int am  = tid >> 1;
    int akh = (tid & 1) << 4;
    const float* abase = A + offA + (long long)(bm + am) * lda + akh;
    // B loader coords
    int btn = tid >> 3;               // TRANSB=1: n-row (32 rows x 8 thr)
    int btk = (tid & 7) << 2;         // TRANSB=1: k quad
    int bnn = tid & 31;               // TRANSB=0: n
    int bnk = (tid >> 5) << 2;        // TRANSB=0: k quad
    const float* bbase = TRANSB
        ? Bp + offB + (long long)(bn + btn) * ldb + btk
        : Bp + offB + (long long)bnk * ldb + (bn + bnn);

    v8f acc0 = {}, acc1 = {};

    for (int k0 = 0; k0 < K; k0 += 32) {
        // ---- A tile: 4 x float4 -> 16 f16 -> 2 x ds_store_b128
        {
            const float4* src = (const float4*)(abase + k0);
            float4 f0 = src[0], f1 = src[1], f2 = src[2], f3 = src[3];
            H8 p0, p1;
            p0.h[0] = (_Float16)f0.x; p0.h[1] = (_Float16)f0.y;
            p0.h[2] = (_Float16)f0.z; p0.h[3] = (_Float16)f0.w;
            p0.h[4] = (_Float16)f1.x; p0.h[5] = (_Float16)f1.y;
            p0.h[6] = (_Float16)f1.z; p0.h[7] = (_Float16)f1.w;
            p1.h[0] = (_Float16)f2.x; p1.h[1] = (_Float16)f2.y;
            p1.h[2] = (_Float16)f2.z; p1.h[3] = (_Float16)f2.w;
            p1.h[4] = (_Float16)f3.x; p1.h[5] = (_Float16)f3.y;
            p1.h[6] = (_Float16)f3.z; p1.h[7] = (_Float16)f3.w;
            *(uint4*)&As[am][akh]     = p0.u;
            *(uint4*)&As[am][akh + 8] = p1.u;
        }
        // ---- B tile into Bs[n][k]
        if (TRANSB) {
            float4 f = *(const float4*)(bbase + k0);
            H4 p;
            p.h[0] = (_Float16)f.x; p.h[1] = (_Float16)f.y;
            p.h[2] = (_Float16)f.z; p.h[3] = (_Float16)f.w;
            *(uint2*)&Bs[btn][btk] = p.u;
        } else {
            // 4 k-values for one n (coalesced across n per k), one b64 DS store
            const float* src = bbase + (long long)k0 * ldb;
            H4 p;
            p.h[0] = (_Float16)src[0];
            p.h[1] = (_Float16)src[(long long)ldb];
            p.h[2] = (_Float16)src[(long long)2 * ldb];
            p.h[3] = (_Float16)src[(long long)3 * ldb];
            *(uint2*)&Bs[bnn][bnk] = p.u;
        }
        __syncthreads();

        // ---- fragments (CDNA5 16-bit layouts)
        V16H a, b0, b1;
        const unsigned* arow = (const unsigned*)(&As[w * 16 + r][0]);
        #pragma unroll
        for (int j = 0; j < 4; ++j) a.u[j]     = arow[j + hf * 4];
        #pragma unroll
        for (int j = 0; j < 4; ++j) a.u[4 + j] = arow[8 + j + hf * 4];
        const unsigned* brow0 = (const unsigned*)(&Bs[r][0]);
        const unsigned* brow1 = (const unsigned*)(&Bs[16 + r][0]);
        #pragma unroll
        for (int j = 0; j < 4; ++j) { b0.u[j] = brow0[j + hf * 4]; b0.u[4 + j] = brow0[8 + j + hf * 4]; }
        #pragma unroll
        for (int j = 0; j < 4; ++j) { b1.u[j] = brow1[j + hf * 4]; b1.u[4 + j] = brow1[8 + j + hf * 4]; }

        acc0 = __builtin_amdgcn_wmma_f32_16x16x32_f16(
            false, a.v, false, b0.v, (short)0, acc0, false, false);
        acc1 = __builtin_amdgcn_wmma_f32_16x16x32_f16(
            false, a.v, false, b1.v, (short)0, acc1, false, false);
        __syncthreads();
    }

    // ---- epilogue: C/D layout: VGPR vv -> M = vv + 8*hf, N = r
    int row0 = bm + w * 16 + hf * 8;
    int col0 = bn + r;
    int col1 = bn + 16 + r;
    #pragma unroll
    for (int vv = 0; vv < 8; ++vv) {
        long long ro = offC + (long long)(row0 + vv) * ldc;
        float v0 = acc0[vv] * alpha;
        float v1 = acc1[vv] * alpha;
        if (bias) { v0 += bias[col0]; v1 += bias[col1]; }
        if (add)  { v0 += add[ro + col0]; v1 += add[ro + col1]; }
        if (act == 1) {
            v0 = 0.5f * v0 * (1.0f + erff(v0 * 0.70710678118f));
            v1 = 0.5f * v1 * (1.0f + erff(v1 * 0.70710678118f));
        } else if (act == 2) {
            v0 = fmaxf(v0, 0.0f);
            v1 = fmaxf(v1, 0.0f);
        }
        C[ro + col0] = v0;
        C[ro + col1] = v1;
    }
}

// ============================================================================
// Row LayerNorm: out = relu?( (x[+addx]-mu)*rsqrt(var+1e-5)*w + b )
// ============================================================================
__global__ __launch_bounds__(256) void layernorm_kernel(
    const float* __restrict__ X, const float* __restrict__ AddX,
    const float* __restrict__ W, const float* __restrict__ Bb,
    float* __restrict__ Out, int cols, int relu)
{
    __shared__ float red[256];
    long long row = blockIdx.x;
    int t = threadIdx.x;
    const float* xr = X + row * cols;
    const float* ar = AddX ? AddX + row * cols : nullptr;
    float vals[8];
    int nv = 0;
    float s = 0.f;
    for (int c = t; c < cols; c += 256) {
        float v = xr[c];
        if (ar) v += ar[c];
        vals[nv++] = v;
        s += v;
    }
    red[t] = s; __syncthreads();
    for (int o = 128; o > 0; o >>= 1) { if (t < o) red[t] += red[t + o]; __syncthreads(); }
    float mu = red[0] / cols;
    __syncthreads();
    float s2 = 0.f;
    for (int i = 0; i < nv; ++i) { float d = vals[i] - mu; s2 += d * d; }
    red[t] = s2; __syncthreads();
    for (int o = 128; o > 0; o >>= 1) { if (t < o) red[t] += red[t + o]; __syncthreads(); }
    float rs = rsqrtf(red[0] / cols + 1e-5f);
    int i = 0;
    for (int c = t; c < cols; c += 256) {
        float v = (vals[i++] - mu) * rs * W[c] + Bb[c];
        if (relu) v = fmaxf(v, 0.f);
        Out[row * cols + c] = v;
    }
}

// ============================================================================
// masked softmax over scores rows [B*NH*S, S]; in place; scale pre-applied
// ============================================================================
__global__ __launch_bounds__(256) void softmax_mask_kernel(
    float* __restrict__ scores, const unsigned char* __restrict__ mask)
{
    __shared__ float red[256];
    long long row = blockIdx.x;
    int b = (int)(row / ((long long)NH_ * S_));
    float* xr = scores + row * S_;
    int t = threadIdx.x;
    int c0 = t, c1 = t + 256;
    float v0 = mask[b * S_ + c0] ? xr[c0] : -1e9f;
    float v1 = mask[b * S_ + c1] ? xr[c1] : -1e9f;
    red[t] = fmaxf(v0, v1); __syncthreads();
    for (int o = 128; o > 0; o >>= 1) { if (t < o) red[t] = fmaxf(red[t], red[t + o]); __syncthreads(); }
    float mx = red[0]; __syncthreads();
    v0 = __expf(v0 - mx); v1 = __expf(v1 - mx);
    red[t] = v0 + v1; __syncthreads();
    for (int o = 128; o > 0; o >>= 1) { if (t < o) red[t] += red[t + o]; __syncthreads(); }
    float inv = 1.0f / red[0];
    xr[c0] = v0 * inv; xr[c1] = v1 * inv;
}

// ============================================================================
// token embedding * sqrt(D) + sinusoidal PE
// ============================================================================
__global__ void embed_kernel(const float* __restrict__ tok_emb,
                             const int* __restrict__ ids,
                             float* __restrict__ h)
{
    int i = blockIdx.x * 256 + threadIdx.x;
    if (i >= B_ * S_ * D_) return;
    int d  = i % D_;
    int bs = i / D_;
    int s  = bs % S_;
    int id = ids[bs];
    float v = tok_emb[(long long)id * D_ + d] * 22.62741699796952f;
    int ii = d & ~1;
    float div = __expf(-(float)ii * (9.210340371976184f / (float)D_));
    float ang = (float)s * div;
    v += (d & 1) ? __cosf(ang) : __sinf(ang);
    h[i] = v;
}

// ============================================================================
// graph node embed: clip, shared-stat LN, per-type affine+Linear+type_emb
// ============================================================================
__global__ __launch_bounds__(256) void graph_embed_kernel(
    const float* __restrict__ x, const int* __restrict__ node_type,
    const float* __restrict__ gln_w, const float* __restrict__ gln_b,
    const float* __restrict__ gw, const float* __restrict__ gb,
    const float* __restrict__ type_emb, float* __restrict__ g)
{
    __shared__ float xs[FG_];
    __shared__ float red[256];
    int n = blockIdx.x, t = threadIdx.x;
    if (t < FG_) {
        float v = x[(long long)n * FG_ + t];
        xs[t] = fminf(fmaxf(v, -10.f), 10.f);
    }
    __syncthreads();
    red[t] = (t < FG_) ? xs[t] : 0.f; __syncthreads();
    for (int o = 128; o > 0; o >>= 1) { if (t < o) red[t] += red[t + o]; __syncthreads(); }
    float mu = red[0] / (float)FG_; __syncthreads();
    float d2 = 0.f;
    if (t < FG_) { float d = xs[t] - mu; d2 = d * d; }
    red[t] = d2; __syncthreads();
    for (int o = 128; o > 0; o >>= 1) { if (t < o) red[t] += red[t + o]; __syncthreads(); }
    float rs = rsqrtf(red[0] / (float)FG_ + 1e-5f);
    int ty = node_type[n];
    const float* wrow = gw + ((long long)ty * HG_ + t) * FG_;
    const float* lw = gln_w + ty * FG_;
    const float* lb = gln_b + ty * FG_;
    float acc = 0.f;
    for (int f = 0; f < FG_; ++f) {
        float xn = (xs[f] - mu) * rs;
        acc += (xn * lw[f] + lb[f]) * wrow[f];
    }
    g[(long long)n * HG_ + t] = acc + gb[ty * HG_ + t] + type_emb[ty * HG_ + t];
}

// ============================================================================
// graph helpers
// ============================================================================
__global__ void fill_kernel(float* __restrict__ p, long long count, float val)
{
    long long i = (long long)blockIdx.x * 256 + threadIdx.x;
    if (i < count) p[i] = val;
}

__global__ void deg_kernel(const int* __restrict__ edge, float* __restrict__ deg)
{
    int e = blockIdx.x * 256 + threadIdx.x;
    if (e < EE_) atomicAdd(&deg[edge[EE_ + e]], 1.0f);
}

__global__ void scatter_kernel(const int* __restrict__ edge,
                               const float* __restrict__ g,
                               float* __restrict__ agg)
{
    long long i = (long long)blockIdx.x * 256 + threadIdx.x;   // E * 64 (4 ch each)
    if (i >= (long long)EE_ * 64) return;
    int e  = (int)(i >> 6);
    int h4 = (int)(i & 63) * 4;
    int s = edge[e], d = edge[EE_ + e];
    const float* gs = g + (long long)s * HG_ + h4;
    float* ad = agg + (long long)d * HG_ + h4;
    atomicAdd(&ad[0], gs[0]);
    atomicAdd(&ad[1], gs[1]);
    atomicAdd(&ad[2], gs[2]);
    atomicAdd(&ad[3], gs[3]);
}

__global__ void divide_kernel(float* __restrict__ agg, const float* __restrict__ deg)
{
    long long i = (long long)blockIdx.x * 256 + threadIdx.x;
    if (i >= (long long)NN_ * HG_) return;
    agg[i] /= fmaxf(deg[i >> 8], 1.0f);   // HG_==256
}

// ============================================================================
// pooling / fusion tail
// ============================================================================
__global__ void pool_log_kernel(const float* __restrict__ y,
                                const unsigned char* __restrict__ mask,
                                float* __restrict__ out)
{
    int i = blockIdx.x * 256 + threadIdx.x;       // B*D
    if (i >= B_ * D_) return;
    int b = i / D_, d = i % D_;
    float s = 0.f, ms = 0.f;
    for (int t = 0; t < S_; ++t) {
        float m = mask[b * S_ + t] ? 1.f : 0.f;
        s  += y[((long long)(b * S_ + t)) * D_ + d] * m;
        ms += m;
    }
    out[i] = s / fmaxf(ms, 1.f);
}

__global__ void pool_graph_kernel(const float* __restrict__ g, float* __restrict__ out)
{
    int h = threadIdx.x;                          // single block of 256
    float s = 0.f;
    for (int n = 0; n < NN_; ++n) s += g[(long long)n * HG_ + h];
    out[h] = s / (float)NN_;
}

__global__ void concat_kernel(const float* __restrict__ lp,
                              const float* __restrict__ gp,
                              float* __restrict__ fin)
{
    int i = blockIdx.x * 256 + threadIdx.x;       // B*FUS
    if (i >= B_ * FUS_) return;
    int b = i / FUS_, c = i % FUS_;
    fin[i] = (c < D_) ? lp[b * D_ + c] : gp[c - D_];
}

__global__ void fusion_mm_kernel(const float* __restrict__ fin,
                                 const float* __restrict__ Wt,
                                 const float* __restrict__ bias,
                                 float* __restrict__ out)
{
    int i = blockIdx.x * 256 + threadIdx.x;       // B*FUS
    if (i >= B_ * FUS_) return;
    int b = i / FUS_, n = i % FUS_;
    float s = bias[n];
    const float* fr = fin + (long long)b * FUS_;
    const float* wr = Wt + (long long)n * FUS_;
    for (int k = 0; k < FUS_; ++k) s += fr[k] * wr[k];
    out[i] = s;
}

// ============================================================================
// host-side launch helper
// ============================================================================
static inline void launch_gemm(hipStream_t stream,
    const float* A, const float* Bp, const float* bias, const float* add, float* C,
    int M, int N, int K, int lda, int ldb, int ldc, int transB, int act, float alpha,
    int batch, int zdiv, long long sA1, long long sA2, long long sB1, long long sB2,
    long long sC1, long long sC2)
{
    dim3 grid(N / 32, M / 128, batch);
    if (transB)
        gemm_wmma_kernel<1><<<grid, 256, 0, stream>>>(A, Bp, bias, add, C, K,
            lda, ldb, ldc, act, alpha, zdiv, sA1, sA2, sB1, sB2, sC1, sC2);
    else
        gemm_wmma_kernel<0><<<grid, 256, 0, stream>>>(A, Bp, bias, add, C, K,
            lda, ldb, ldc, act, alpha, zdiv, sA1, sA2, sB1, sB2, sC1, sC2);
}

extern "C" void kernel_launch(void* const* d_in, const int* in_sizes, int n_in,
                              void* d_out, int out_size, void* d_ws, size_t ws_size,
                              hipStream_t stream)
{
    (void)in_sizes; (void)n_in; (void)out_size; (void)ws_size;

    const float* tok_emb = (const float*)d_in[0];
    const float* qkv_w   = (const float*)d_in[1];
    const float* qkv_b   = (const float*)d_in[2];
    const float* out_w   = (const float*)d_in[3];
    const float* out_b   = (const float*)d_in[4];
    const float* ln1_w   = (const float*)d_in[5];
    const float* ln1_b   = (const float*)d_in[6];
    const float* ln2_w   = (const float*)d_in[7];
    const float* ln2_b   = (const float*)d_in[8];
    const float* ff1_w   = (const float*)d_in[9];
    const float* ff1_b   = (const float*)d_in[10];
    const float* ff2_w   = (const float*)d_in[11];
    const float* ff2_b   = (const float*)d_in[12];
    const float* fln_w   = (const float*)d_in[13];
    const float* fln_b   = (const float*)d_in[14];
    const float* xg      = (const float*)d_in[15];
    const float* gln_w   = (const float*)d_in[16];
    const float* gln_b   = (const float*)d_in[17];
    const float* gw      = (const float*)d_in[18];
    const float* gb      = (const float*)d_in[19];
    const float* type_e  = (const float*)d_in[20];
    const float* sage_wl = (const float*)d_in[21];
    const float* sage_bl = (const float*)d_in[22];
    const float* sage_wr = (const float*)d_in[23];
    const float* gnorm_w = (const float*)d_in[24];
    const float* gnorm_b = (const float*)d_in[25];
    const float* fusion_w    = (const float*)d_in[26];
    const float* fusion_b    = (const float*)d_in[27];
    const float* fusion_ln_w = (const float*)d_in[28];
    const float* fusion_ln_b = (const float*)d_in[29];
    const int*           input_ids = (const int*)d_in[30];
    const unsigned char* amask     = (const unsigned char*)d_in[31];
    const int*           edge      = (const int*)d_in[32];
    const int*           node_type = (const int*)d_in[33];

    float* ws = (float*)d_ws;
    const long long GSZ = (long long)NP_ * HG_;   // 12,812,288
    // ---- graph-phase layout (used first, then reused) ----
    float* g     = ws + 0;
    float* agg   = ws + 12812288;
    float* t1    = ws + 25624576;
    float* t2    = ws + 38436864;
    float* deg   = ws + 51249152;      // NN
    float* gpool = ws + 51299200;      // HG
    float* lpool = ws + 51299584;      // B*D
    float* finb  = ws + 51303680;      // B*FUS
    float* ftmp  = ws + 51309824;      // B*FUS
    // ---- transformer layout (reuses [0 .. 37.8M) after graph phase) ----
    float* h   = ws + 0;               // B*S*D  = 2,097,152
    float* y   = ws + 2097152;         // B*S*D
    float* qkv = ws + 4194304;         // B*S*3D = 6,291,456
    float* sc  = ws + 10485760;        // B*NH*S*S = 16,777,216
    float* ob  = ws + 27262976;        // B*S*D
    float* ffb = ws + 29360128;        // B*S*FF = 8,388,608

    const int BS = B_ * S_;            // 4096 token rows (multiple of 128)

    // ======================= graph encoder =======================
    // zero the pad rows of g once so guard-free GEMM reads zeros there
    fill_kernel<<<(int)(((long long)(NP_ - NN_) * HG_ + 255) / 256), 256, 0, stream>>>(
        g + (long long)NN_ * HG_, (long long)(NP_ - NN_) * HG_, 0.f);
    graph_embed_kernel<<<NN_, 256, 0, stream>>>(xg, node_type, gln_w, gln_b,
                                                gw, gb, type_e, g);
    fill_kernel<<<(NN_ + 255) / 256, 256, 0, stream>>>(deg, NN_, 0.f);
    deg_kernel<<<(EE_ + 255) / 256, 256, 0, stream>>>(edge, deg);

    for (int l = 0; l < 2; ++l) {
        fill_kernel<<<(int)((GSZ + 255) / 256), 256, 0, stream>>>(agg, GSZ, 0.f);
        scatter_kernel<<<(int)(((long long)EE_ * 64 + 255) / 256), 256, 0, stream>>>(
            edge, g, agg);
        divide_kernel<<<(int)(((long long)NN_ * HG_ + 255) / 256), 256, 0, stream>>>(
            agg, deg);
        // t1 = agg @ Wl^T + bl
        launch_gemm(stream, agg, sage_wl + (long long)l * HG_ * HG_,
                    sage_bl + l * HG_, nullptr, t1,
                    NP_, HG_, HG_, HG_, HG_, HG_, 1, 0, 1.0f,
                    1, 1, 0, 0, 0, 0, 0, 0);
        // t2 = g @ Wr^T + t1
        launch_gemm(stream, g, sage_wr + (long long)l * HG_ * HG_,
                    nullptr, t1, t2,
                    NP_, HG_, HG_, HG_, HG_, HG_, 1, 0, 1.0f,
                    1, 1, 0, 0, 0, 0, 0, 0);
        // g = relu(ln(t2 + g))   (pad rows of g untouched -> stay zero)
        layernorm_kernel<<<NN_, 256, 0, stream>>>(t2, g, gnorm_w + l * HG_,
                                                  gnorm_b + l * HG_, g, HG_, 1);
    }
    pool_graph_kernel<<<1, 256, 0, stream>>>(g, gpool);

    // ======================= transformer =======================
    embed_kernel<<<(B_ * S_ * D_ + 255) / 256, 256, 0, stream>>>(tok_emb, input_ids, h);

    for (int l = 0; l < L_; ++l) {
        // y = ln1(h)
        layernorm_kernel<<<BS, 256, 0, stream>>>(h, nullptr, ln1_w + l * D_,
                                                 ln1_b + l * D_, y, D_, 0);
        // qkv = y @ qkv_w[l]^T + qkv_b[l]
        launch_gemm(stream, y, qkv_w + (long long)l * 3 * D_ * D_,
                    qkv_b + l * 3 * D_, nullptr, qkv,
                    BS, 3 * D_, D_, D_, D_, 3 * D_, 1, 0, 1.0f,
                    1, 1, 0, 0, 0, 0, 0, 0);
        // scores[b,hd] = (Q @ K^T) / sqrt(DH)     (batched over B*NH)
        launch_gemm(stream, qkv, qkv + D_, nullptr, nullptr, sc,
                    S_, S_, DH_, 3 * D_, 3 * D_, S_, 1, 0, 0.125f,
                    B_ * NH_, NH_,
                    (long long)S_ * 3 * D_, DH_,
                    (long long)S_ * 3 * D_, DH_,
                    (long long)NH_ * S_ * S_, (long long)S_ * S_);
        // softmax with mask
        softmax_mask_kernel<<<B_ * NH_ * S_, 256, 0, stream>>>(sc, amask);
        // o[b,s,hd*64..] = attn @ V                 (batched, writes [B,S,D])
        launch_gemm(stream, sc, qkv + 2 * D_, nullptr, nullptr, ob,
                    S_, DH_, S_, S_, 3 * D_, D_, 0, 0, 1.0f,
                    B_ * NH_, NH_,
                    (long long)NH_ * S_ * S_, (long long)S_ * S_,
                    (long long)S_ * 3 * D_, DH_,
                    (long long)S_ * D_, DH_);
        // h = h + o @ out_w^T + out_b
        launch_gemm(stream, ob, out_w + (long long)l * D_ * D_,
                    out_b + l * D_, h, h,
                    BS, D_, D_, D_, D_, D_, 1, 0, 1.0f,
                    1, 1, 0, 0, 0, 0, 0, 0);
        // y = ln2(h)
        layernorm_kernel<<<BS, 256, 0, stream>>>(h, nullptr, ln2_w + l * D_,
                                                 ln2_b + l * D_, y, D_, 0);
        // ffb = gelu(y @ ff1^T + b1)
        launch_gemm(stream, y, ff1_w + (long long)l * FF_ * D_,
                    ff1_b + l * FF_, nullptr, ffb,
                    BS, FF_, D_, D_, D_, FF_, 1, 1, 1.0f,
                    1, 1, 0, 0, 0, 0, 0, 0);
        // h = h + ffb @ ff2^T + b2
        launch_gemm(stream, ffb, ff2_w + (long long)l * D_ * FF_,
                    ff2_b + l * D_, h, h,
                    BS, D_, FF_, FF_, FF_, D_, 1, 0, 1.0f,
                    1, 1, 0, 0, 0, 0, 0, 0);
    }
    // y = final ln(h)
    layernorm_kernel<<<BS, 256, 0, stream>>>(h, nullptr, fln_w, fln_b, y, D_, 0);
    pool_log_kernel<<<(B_ * D_ + 255) / 256, 256, 0, stream>>>(y, amask, lpool);

    // ======================= fusion =======================
    concat_kernel<<<(B_ * FUS_ + 255) / 256, 256, 0, stream>>>(lpool, gpool, finb);
    fusion_mm_kernel<<<(B_ * FUS_ + 255) / 256, 256, 0, stream>>>(finb, fusion_w,
                                                                  fusion_b, ftmp);
    layernorm_kernel<<<B_, 256, 0, stream>>>(ftmp, nullptr, fusion_ln_w, fusion_ln_b,
                                             (float*)d_out, FUS_, 1);
}
